// PatchedQWenAttention_23459111370958
// MI455X (gfx1250) — matrix operand order repaired and checked by reading, not compiled, not run on hardware
//
#include <hip/hip_runtime.h>

// ---------------------------------------------------------------------------
// PatchedQWenAttention for MI455X (gfx1250, wave32, WMMA bf16, async-to-LDS)
// B=8 QLEN=256 HIDDEN=4096 HEADS=32 HDIM=128 KV=768 T=2048
// ---------------------------------------------------------------------------

typedef __attribute__((ext_vector_type(16))) __bf16 v16bf;
typedef __attribute__((ext_vector_type(8)))  float  v8f;

union Frag {
  uint4 u[2];
  v16bf v;
};

__device__ __forceinline__ unsigned short f2bf(float f) {
  unsigned int u = __float_as_uint(f);
  u += 0x7FFFu + ((u >> 16) & 1u);         // round-to-nearest-even
  return (unsigned short)(u >> 16);
}

// LDS byte offset of a __shared__ object (addrspace(3) integer value).
__device__ __forceinline__ unsigned int lds_addr(const void* p) {
  return (unsigned int)(size_t)(const __attribute__((address_space(3))) void*)p;
}

// One 16-byte async DMA: global -> LDS, tracked by ASYNCcnt.
__device__ __forceinline__ void async_copy16(unsigned int lds_off, const void* g) {
  asm volatile("global_load_async_to_lds_b128 %0, %1, off"
               :: "v"(lds_off), "v"(g) : "memory");
}

__device__ __forceinline__ void wait_async0() {
#if __has_builtin(__builtin_amdgcn_s_wait_asynccnt)
  __builtin_amdgcn_s_wait_asynccnt(0);
#else
  asm volatile("s_wait_asynccnt 0x0" ::: "memory");
#endif
}

// ---------------------------------------------------------------------------
// GEMM: C[M,N] = A[M,K] * W[N,K]^T (+bias), f32 in/out, bf16 WMMA math.
// Block = 256 threads = 8 waves; tile 128x128; wave tile 32x64.
// ---------------------------------------------------------------------------
#define ASTR 40   // LDS row stride (elements): 80B, 16B-aligned, conflict-free

__global__ __launch_bounds__(256) void gemm_xwT_bf16(
    const float* __restrict__ A, const float* __restrict__ W,
    const float* __restrict__ bias, float* __restrict__ C,
    int M, int N, int K)
{
  __shared__ __align__(16) unsigned short sA[128][ASTR];
  __shared__ __align__(16) unsigned short sB[128][ASTR];

  const int tid   = threadIdx.x;
  const int lane  = tid & 31;
  const int wave  = tid >> 5;
  const int wm    = wave >> 1;      // 0..3 (M)
  const int wn    = wave & 1;       // 0..1 (N)
  const int l15   = lane & 15;
  const int lhalf = lane >> 4;
  const int kbA   = lhalf * 8;      // A-frag: halves hold K {0..7,16..23}/{8..15,24..31}
  const int kbB   = lhalf * 8;      // B-frag: halves hold K {0..7,16..23}/{8..15,24..31}
  const int bm    = blockIdx.x * 128;
  const int bn    = blockIdx.y * 128;

  v8f acc[2][4];
  for (int i = 0; i < 2; ++i)
    for (int j = 0; j < 4; ++j)
      for (int e = 0; e < 8; ++e) acc[i][j][e] = 0.0f;

  for (int k0 = 0; k0 < K; k0 += 32) {
    __syncthreads();
    for (int it = 0; it < 4; ++it) {
      int idx = tid + it * 256;            // 1024 float4 slots per tile
      int row = idx >> 3;                  // 0..127
      int c4  = (idx & 7) * 4;             // 0..28
      const float* pa = &A[(size_t)(bm + row) * K + k0 + c4];
      const float* pb = &W[(size_t)(bn + row) * K + k0 + c4];
      float4 a = *(const float4*)pa;
      float4 b = *(const float4*)pb;
      sA[row][c4 + 0] = f2bf(a.x); sA[row][c4 + 1] = f2bf(a.y);
      sA[row][c4 + 2] = f2bf(a.z); sA[row][c4 + 3] = f2bf(a.w);
      sB[row][c4 + 0] = f2bf(b.x); sB[row][c4 + 1] = f2bf(b.y);
      sB[row][c4 + 2] = f2bf(b.z); sB[row][c4 + 3] = f2bf(b.w);
      __builtin_prefetch(pa + 32, 0, 0);   // next K tile -> global_prefetch_b8
      __builtin_prefetch(pb + 32, 0, 0);
    }
    __syncthreads();

    Frag fa[2], fb[4];
    for (int sm = 0; sm < 2; ++sm) {
      int r = wm * 32 + sm * 16 + l15;
      fa[sm].u[0] = *(const uint4*)&sA[r][kbA];
      fa[sm].u[1] = *(const uint4*)&sA[r][16 + kbA];
    }
    for (int sn = 0; sn < 4; ++sn) {
      int r = wn * 64 + sn * 16 + l15;
      fb[sn].u[0] = *(const uint4*)&sB[r][kbB];
      fb[sn].u[1] = *(const uint4*)&sB[r][16 + kbB];
    }
    for (int sm = 0; sm < 2; ++sm)
      for (int sn = 0; sn < 4; ++sn)
        acc[sm][sn] = __builtin_amdgcn_wmma_f32_16x16x32_bf16(
            false, fa[sm].v, false, fb[sn].v, (short)0, acc[sm][sn],
            false, false);
  }

  for (int sm = 0; sm < 2; ++sm)
    for (int sn = 0; sn < 4; ++sn) {
      int gn = bn + wn * 64 + sn * 16 + l15;
      float bv = bias ? bias[gn] : 0.0f;
      for (int r = 0; r < 8; ++r) {
        int gm = bm + wm * 32 + sm * 16 + r + 8 * lhalf;  // C/D row layout
        C[(size_t)gm * N + gn] = acc[sm][sn][r] + bv;
      }
    }
}

// ---------------------------------------------------------------------------
// Gather paged KV cache -> dense bf16. K as [b,h,kv,d]; V pre-transposed as
// [b,h,d,kv] so the attention kernel can async-DMA both without reshaping.
// ---------------------------------------------------------------------------
__global__ __launch_bounds__(256) void gather_kv(
    const float* __restrict__ k_cache, const float* __restrict__ v_cache,
    const int* __restrict__ block_offsets,
    unsigned short* __restrict__ Kg, unsigned short* __restrict__ Vt)
{
  int idx = blockIdx.x * 256 + threadIdx.x;     // 8*768*32*(128/4)
  int d0  = (idx & 31) * 4;
  int t   = idx >> 5;
  int h   = t & 31;  t >>= 5;
  int kv  = t % 768;
  int b   = t / 768;
  int bh  = b * 32 + h;

  int blk  = block_offsets[b * 12 + (kv >> 6)];
  int slot = kv & 63;
  size_t src = (((size_t)blk * 64 + slot) * 32 + h) * 128 + d0;

  float4 kk = *(const float4*)&k_cache[src];
  float4 vv = *(const float4*)&v_cache[src];
  union { unsigned short s[4]; uint2 u; } pk;
  pk.s[0] = f2bf(kk.x); pk.s[1] = f2bf(kk.y);
  pk.s[2] = f2bf(kk.z); pk.s[3] = f2bf(kk.w);
  *(uint2*)&Kg[((size_t)bh * 768 + kv) * 128 + d0] = pk.u;

  Vt[((size_t)bh * 128 + d0 + 0) * 768 + kv] = f2bf(vv.x);
  Vt[((size_t)bh * 128 + d0 + 1) * 768 + kv] = f2bf(vv.y);
  Vt[((size_t)bh * 128 + d0 + 2) * 768 + kv] = f2bf(vv.z);
  Vt[((size_t)bh * 128 + d0 + 3) * 768 + kv] = f2bf(vv.w);
}

// ---------------------------------------------------------------------------
// RoPE q/k + scatter new tokens into gathered K/Vt + emit roped Q (bf16)
// one thread per (token, head, pair j<64)
// ---------------------------------------------------------------------------
__global__ __launch_bounds__(256) void rope_scatter(
    const float* __restrict__ qkv, const int* __restrict__ hist,
    unsigned short* __restrict__ Qr, unsigned short* __restrict__ Kg,
    unsigned short* __restrict__ Vt)
{
  int idx = blockIdx.x * 256 + threadIdx.x;   // 2048*32*64
  int j  = idx & 63;
  int h  = (idx >> 6) & 31;
  int t  = idx >> 11;
  int b  = t >> 8;
  int qp = t & 255;
  int bh = b * 32 + h;
  int pos = hist[b] + qp;

  float inv_freq = __expf(-(float)j * (2.0f / 128.0f) * logf(10000.0f));
  float ang = (float)pos * inv_freq;
  float cs = cosf(ang), sn = sinf(ang);

  size_t base = (size_t)t * 12288 + h * 128 + j;
  float q1 = qkv[base],        q2 = qkv[base + 64];
  float k1 = qkv[base + 4096], k2 = qkv[base + 4096 + 64];
  float v1 = qkv[base + 8192], v2 = qkv[base + 8192 + 64];
  float rq1 = q1 * cs - q2 * sn, rq2 = q2 * cs + q1 * sn;
  float rk1 = k1 * cs - k2 * sn, rk2 = k2 * cs + k1 * sn;

  size_t qb = ((size_t)bh * 256 + qp) * 128 + j;
  Qr[qb] = f2bf(rq1); Qr[qb + 64] = f2bf(rq2);
  size_t kb = ((size_t)bh * 768 + pos) * 128 + j;
  Kg[kb] = f2bf(rk1); Kg[kb + 64] = f2bf(rk2);
  Vt[((size_t)bh * 128 + j)      * 768 + pos] = f2bf(v1);
  Vt[((size_t)bh * 128 + j + 64) * 768 + pos] = f2bf(v2);
}

// ---------------------------------------------------------------------------
// Flash attention: block = (b, h, qtile of 128); 8 waves x 16 q-rows.
// Double-buffered async-to-LDS K/V staging overlapped with WMMA compute.
// ---------------------------------------------------------------------------
__global__ __launch_bounds__(256) void flash_attn(
    const unsigned short* __restrict__ Qr, const unsigned short* __restrict__ Kg,
    const unsigned short* __restrict__ Vt, const int* __restrict__ hist,
    float* __restrict__ attn)
{
  __shared__ __align__(16) unsigned short sK[2][64][136];   // K chunk  [kv][d]
  __shared__ __align__(16) unsigned short sVt[2][128][72];  // V chunk  [d][kv]
  __shared__ __align__(16) unsigned short sP[8][16][72];    // probs    [m][kv]

  const int tid   = threadIdx.x;
  const int lane  = tid & 31;
  const int wave  = tid >> 5;
  const int l15   = lane & 15;
  const int lhalf = lane >> 4;
  const int kbA   = lhalf * 8;
  const int kbB   = lhalf * 8;

  const int qt = blockIdx.x & 1;
  const int bh = blockIdx.x >> 1;
  const int h  = bh & 31;
  const int b  = bh >> 5;
  const int hb = hist[b];

  const unsigned short* Qh = Qr + (size_t)bh * 256 * 128;
  const unsigned short* Kh = Kg + (size_t)bh * 768 * 128;
  const unsigned short* Vh = Vt + (size_t)bh * 128 * 768;

  // Q fragments for this wave's 16 rows, 4 d-windows of 32
  Frag qf[4];
  {
    const unsigned short* qp = Qh + (size_t)(qt * 128 + wave * 16 + l15) * 128;
    for (int kc = 0; kc < 4; ++kc) {
      qf[kc].u[0] = *(const uint4*)&qp[kc * 32 + kbA];
      qf[kc].u[1] = *(const uint4*)&qp[kc * 32 + 16 + kbA];
    }
  }

  v8f oacc[8];
  for (int ds = 0; ds < 8; ++ds)
    for (int e = 0; e < 8; ++e) oacc[ds][e] = 0.0f;
  float mrun[8], lrun[8];
  for (int r = 0; r < 8; ++r) { mrun[r] = -3.0e30f; lrun[r] = 0.0f; }

  const int qmaxg = hb + qt * 128 + 127;       // largest visible kv column
  const int cmax  = min(11, qmaxg >> 6);       // last non-masked chunk
  const float scale = 0.08838834764831845f;    // 1/sqrt(128)

  // Async stage of one 64-wide kv chunk (this thread's share: 4+4 x 16B DMAs)
  auto stage = [&](int c, int buf) {
    for (int it = 0; it < 4; ++it) {
      int idx = tid + it * 256;                // 1024 x 16B (K)
      int kk  = idx >> 4;
      int g   = (idx & 15) * 8;
      async_copy16(lds_addr(&sK[buf][kk][g]),
                   &Kh[(size_t)(c * 64 + kk) * 128 + g]);
    }
    for (int it = 0; it < 4; ++it) {
      int idx = tid + it * 256;                // 1024 x 16B (V, transposed)
      int d   = idx >> 3;
      int g   = (idx & 7) * 8;
      async_copy16(lds_addr(&sVt[buf][d][g]),
                   &Vh[(size_t)d * 768 + c * 64 + g]);
    }
  };

  stage(0, 0);
  wait_async0();

  for (int c = 0; c <= cmax; ++c) {
    const int buf = c & 1;
    __syncthreads();                 // publish staged chunk; retire old reads
    if (c < cmax) stage(c + 1, buf ^ 1);   // overlap DMA with compute

    // ---- S = Q K^T : 4 subtiles of 16 kv columns ----
    v8f sacc[4];
    for (int ns = 0; ns < 4; ++ns) {
      for (int e = 0; e < 8; ++e) sacc[ns][e] = 0.0f;
      int krow = ns * 16 + l15;                // B-frag: lane = kv column
      for (int kc = 0; kc < 4; ++kc) {
        Frag kb;
        kb.u[0] = *(const uint4*)&sK[buf][krow][kc * 32 + kbB];
        kb.u[1] = *(const uint4*)&sK[buf][krow][kc * 32 + 16 + kbB];
        sacc[ns] = __builtin_amdgcn_wmma_f32_16x16x32_bf16(
            false, qf[kc].v, false, kb.v, (short)0, sacc[ns], false, false);
      }
    }

    // ---- causal mask + online softmax (row = (r, lane-half)) ----
    for (int r = 0; r < 8; ++r) {
      int qg = hb + qt * 128 + wave * 16 + r + 8 * lhalf;
      float sv[4];
      float mx = -3.0e30f;
      for (int ns = 0; ns < 4; ++ns) {
        int col = c * 64 + ns * 16 + l15;
        float s = sacc[ns][r] * scale;
        if (col > qg) s = -3.0e30f;
        sv[ns] = s;
        mx = fmaxf(mx, s);
      }
      for (int off = 1; off < 16; off <<= 1)
        mx = fmaxf(mx, __shfl_xor(mx, off, 32));
      float mnew  = fmaxf(mrun[r], mx);
      float alpha = __expf(mrun[r] - mnew);
      float rsum = 0.0f;
      for (int ns = 0; ns < 4; ++ns) {
        float p = __expf(sv[ns] - mnew);
        rsum += p;
        sP[wave][r + 8 * lhalf][ns * 16 + l15] = f2bf(p);
      }
      for (int off = 1; off < 16; off <<= 1)
        rsum += __shfl_xor(rsum, off, 32);
      lrun[r] = lrun[r] * alpha + rsum;
      mrun[r] = mnew;
      for (int ds = 0; ds < 8; ++ds) oacc[ds][r] *= alpha;
    }
    __syncthreads();                           // sP visible wave-wide

    // ---- O += P V ----
    Frag pf[2];
    for (int ks = 0; ks < 2; ++ks) {
      pf[ks].u[0] = *(const uint4*)&sP[wave][l15][ks * 32 + kbA];
      pf[ks].u[1] = *(const uint4*)&sP[wave][l15][ks * 32 + 16 + kbA];
    }
    for (int ds = 0; ds < 8; ++ds) {
      int drow = ds * 16 + l15;                // B-frag: lane = d column
      for (int ks = 0; ks < 2; ++ks) {
        Frag vb;
        vb.u[0] = *(const uint4*)&sVt[buf][drow][ks * 32 + kbB];
        vb.u[1] = *(const uint4*)&sVt[buf][drow][ks * 32 + 16 + kbB];
        oacc[ds] = __builtin_amdgcn_wmma_f32_16x16x32_bf16(
            false, pf[ks].v, false, vb.v, (short)0, oacc[ds], false, false);
      }
    }

    wait_async0();                 // next chunk resident before next compute
  }

  // ---- normalize and write attn output (token-major, heads concatenated) --
  for (int r = 0; r < 8; ++r) {
    int trow = b * 256 + qt * 128 + wave * 16 + r + 8 * lhalf;
    float inv = 1.0f / lrun[r];
    for (int ds = 0; ds < 8; ++ds) {
      int d = ds * 16 + l15;
      attn[(size_t)trow * 4096 + h * 128 + d] = oacc[ds][r] * inv;
    }
  }
}

// ---------------------------------------------------------------------------
extern "C" void kernel_launch(void* const* d_in, const int* in_sizes, int n_in,
                              void* d_out, int out_size, void* d_ws, size_t ws_size,
                              hipStream_t stream) {
  const float* x        = (const float*)d_in[0];  // (1, 2048, 4096)
  const float* c_attn_w = (const float*)d_in[1];  // (12288, 4096)
  const float* c_attn_b = (const float*)d_in[2];  // (12288,)
  const float* c_proj_w = (const float*)d_in[3];  // (4096, 4096)
  const float* k_cache  = (const float*)d_in[4];  // (96,64,32,128)
  const float* v_cache  = (const float*)d_in[5];
  const int*   hist     = (const int*)d_in[6];    // (8,)
  const int*   blkoff   = (const int*)d_in[7];    // (8,12)
  float*       out      = (float*)d_out;          // (1, 2048, 4096)

  char* ws = (char*)d_ws;
  float*          qkv  = (float*)(ws);                             // 100663296 B
  float*          attn = (float*)(ws + 100663296u);                //  33554432 B
  unsigned short* Qr   = (unsigned short*)(ws + 134217728u);       //  16777216 B
  unsigned short* Kg   = (unsigned short*)(ws + 150994944u);       //  50331648 B
  unsigned short* Vt   = (unsigned short*)(ws + 201326592u);       //  50331648 B

  // 1) QKV projection (bf16 WMMA, f32 accumulate, +bias)
  gemm_xwT_bf16<<<dim3(16, 96), 256, 0, stream>>>(
      x, c_attn_w, c_attn_b, qkv, 2048, 12288, 4096);
  // 2) gather paged cache -> dense bf16 K / transposed V
  gather_kv<<<24576, 256, 0, stream>>>(k_cache, v_cache, blkoff, Kg, Vt);
  // 3) RoPE + scatter new tokens + roped Q
  rope_scatter<<<16384, 256, 0, stream>>>(qkv, hist, Qr, Kg, Vt);
  // 4) flash attention (bf16 WMMA; async-to-LDS double-buffered staging)
  flash_attn<<<512, 256, 0, stream>>>(Qr, Kg, Vt, hist, attn);
  // 5) output projection
  gemm_xwT_bf16<<<dim3(16, 32), 256, 0, stream>>>(
      attn, c_proj_w, nullptr, out, 2048, 4096, 4096);
}
